// MultiHeadAttention_44143673868309
// MI455X (gfx1250) — matrix operand order repaired and checked by reading, not compiled, not run on hardware
//
#include <hip/hip_runtime.h>

// ---------------------------------------------------------------------------
// MI455X (gfx1250) multi-head attention, wave32 + WMMA f16 (f32 accumulate).
// B=2, NQ=SK=2048, D=1024, H=16, DK=DV=64.  SCALE = sqrt(64) = 8 (multiplied).
//
// Memory-bound on the attn_weights (536MB) + mask (134MB) single-use stream:
// those loads are non-temporal + prefetched; Q/K/V/O intermediates are f16 in
// workspace (32MB, L2-resident).  V is stored pre-transposed [b][h][dv][sk].
// K/V tiles stream into ping-pong LDS buffers via ASYNC loads (ASYNCcnt)
// when the toolchain exposes them, else register double-buffering.
// ---------------------------------------------------------------------------

typedef __attribute__((ext_vector_type(8)))  _Float16 v8h;
typedef __attribute__((ext_vector_type(16))) _Float16 v16h;
typedef __attribute__((ext_vector_type(8)))  float    v8f;
typedef __attribute__((ext_vector_type(4)))  float    v4f;

union AB16 { v16h v; v8h h[2]; };

#define WMMA16(A, Bm, C) \
    __builtin_amdgcn_wmma_f32_16x16x32_f16(false, (A), false, (Bm), (short)0, (C), false, false)

#if __has_builtin(__builtin_amdgcn_global_load_async_to_lds_b128) && \
    __has_builtin(__builtin_amdgcn_s_wait_asynccnt)
#define HAVE_ASYNC_LDS 1
#else
#define HAVE_ASYNC_LDS 0
#endif

constexpr int   BATCH  = 2;
constexpr int   NQ     = 2048;
constexpr int   SK     = 2048;
constexpr int   DMODEL = 1024;
constexpr int   H      = 16;
constexpr int   HD     = 1024;   // H * 64
constexpr float SCALE  = 8.0f;   // sqrt(DK), reference MULTIPLIES

__device__ __forceinline__ v4f nt_load4(const float* p) {
    return __builtin_nontemporal_load((const v4f*)p);
}
__device__ __forceinline__ unsigned long long nt_load_u64(const unsigned char* p) {
    return __builtin_nontemporal_load((const unsigned long long*)p);
}

#if HAVE_ASYNC_LDS
// Builtin signature (from compiler diagnostic): param 0 is
// "int __attribute__((vector_size(16))) addrspace(1)*" (printed '__device__').
typedef int v4i_vs __attribute__((vector_size(16)));
typedef __attribute__((address_space(1))) v4i_vs* g_v4i_ptr;
typedef __attribute__((address_space(3))) v4i_vs* l_v4i_ptr;

__device__ __forceinline__ void async_cp16(const _Float16* gsrc, _Float16* ldst) {
    // generic->global is value-identity; LDS address is addr[31:0] per ISA.
    __builtin_amdgcn_global_load_async_to_lds_b128(
        (g_v4i_ptr)(unsigned long long)gsrc,
        (l_v4i_ptr)(unsigned)(unsigned long long)ldst, 0, 0);
}
#endif

// ---------------------------------------------------------------------------
// Tiled GEMM: Y[M][N] = A[M][K] * W[K][N] (+ bias), 128x128 block tile, 8
// waves; each wave computes a 16x128 strip (8 WMMA per 32-wide K step).
// MODE 0: A f32 -> Y f16 row-major            (Q/K projections)
// MODE 1: A f32 -> Y f16 transposed [b][h][dv][sk]  (V projection)
// MODE 2: A f16 -> Y f32 + bias, non-temporal store (output projection)
// ---------------------------------------------------------------------------
template <int MODE>
__global__ __launch_bounds__(256)
void gemm_wmma_kernel(const void* __restrict__ Aptr,
                      const float* __restrict__ W,
                      const float* __restrict__ bias,
                      void* __restrict__ Yptr,
                      int M, int N, int K)
{
    __shared__ _Float16 As[128][40];   // 128 rows x 32 k (pad 8)
    __shared__ _Float16 Bt[128][40];   // 128 cols x 32 k (transposed, pad 8)

    const int rowBase = blockIdx.y * 128;
    const int colBase = blockIdx.x * 128;
    const int t    = threadIdx.x;
    const int w    = t >> 5;
    const int lane = t & 31;
    const int lm   = lane & 15;
    const bool hi  = lane >= 16;

    v8f acc[8];
#pragma unroll
    for (int j = 0; j < 8; ++j)
#pragma unroll
        for (int i = 0; i < 8; ++i) acc[j][i] = 0.0f;

    for (int kt = 0; kt < K; kt += 32) {
        // ---- stage A tile: 128 x 32 (f16 in LDS) ----
        {
            const int ar = t >> 1;            // 0..127
            const int ac = (t & 1) * 16;      // 0 or 16
            if (MODE == 2) {
                const _Float16* ap = (const _Float16*)Aptr +
                                     (size_t)(rowBase + ar) * K + kt + ac;
                *(v8h*)&As[ar][ac]     = *(const v8h*)(ap);
                *(v8h*)&As[ar][ac + 8] = *(const v8h*)(ap + 8);
            } else {
                const float* ap = (const float*)Aptr +
                                  (size_t)(rowBase + ar) * K + kt + ac;
                float tmp[16];
                *(v4f*)(tmp + 0)  = *(const v4f*)(ap + 0);
                *(v4f*)(tmp + 4)  = *(const v4f*)(ap + 4);
                *(v4f*)(tmp + 8)  = *(const v4f*)(ap + 8);
                *(v4f*)(tmp + 12) = *(const v4f*)(ap + 12);
                v8h h0, h1;
#pragma unroll
                for (int i = 0; i < 8; ++i) {
                    h0[i] = (_Float16)tmp[i];
                    h1[i] = (_Float16)tmp[8 + i];
                }
                *(v8h*)&As[ar][ac]     = h0;
                *(v8h*)&As[ar][ac + 8] = h1;
            }
        }
        // ---- stage W tile transposed: Bt[n][k], 32 x 128 source ----
        {
            const int bk = t >> 3;            // 0..31
            const int bn = (t & 7) * 16;      // 0..112
            const float* wp = W + (size_t)(kt + bk) * N + colBase + bn;
            float wt[16];
            *(v4f*)(wt + 0)  = *(const v4f*)(wp + 0);
            *(v4f*)(wt + 4)  = *(const v4f*)(wp + 4);
            *(v4f*)(wt + 8)  = *(const v4f*)(wp + 8);
            *(v4f*)(wt + 12) = *(const v4f*)(wp + 12);
#pragma unroll
            for (int i = 0; i < 16; ++i) Bt[bn + i][bk] = (_Float16)wt[i];
        }
        __syncthreads();

        // ---- A fragment: lane = M row, split-K halves per CDNA5 layout ----
        const int offA = hi ? 8 : 0;
        AB16 af;
        af.h[0] = *(const v8h*)&As[16 * w + lm][offA];
        af.h[1] = *(const v8h*)&As[16 * w + lm][offA + 16];

        const int offB = hi ? 8 : 0;
#pragma unroll
        for (int j = 0; j < 8; ++j) {
            AB16 bf;
            bf.h[0] = *(const v8h*)&Bt[16 * j + lm][offB];
            bf.h[1] = *(const v8h*)&Bt[16 * j + lm][offB + 16];
            acc[j] = WMMA16(af.v, bf.v, acc[j]);
        }
        __syncthreads();
    }

    // ---- store: C layout lane = N col, VGPR r -> M = r + 8*hi ----
#pragma unroll
    for (int j = 0; j < 8; ++j) {
        const int col = colBase + 16 * j + lm;
#pragma unroll
        for (int r = 0; r < 8; ++r) {
            const int row = rowBase + 16 * w + r + (hi ? 8 : 0);
            if (MODE == 0) {
                ((_Float16*)Yptr)[(size_t)row * N + col] = (_Float16)acc[j][r];
            } else if (MODE == 1) {
                // V transposed: row = b*SK + sk, col = h*64 + dv
                const int b  = row >> 11;          // /2048
                const int sk = row & 2047;
                ((_Float16*)Yptr)[((size_t)((b << 4) | (col >> 6)) * 64 +
                                   (col & 63)) * SK + sk] = (_Float16)acc[j][r];
            } else {
                __builtin_nontemporal_store(acc[j][r] + bias[col],
                                            (float*)Yptr + (size_t)row * N + col);
            }
        }
    }
}

// ---------------------------------------------------------------------------
// Flash attention. Block = (b, h, 128 q rows), 8 waves x 16-row q tiles,
// 64 keys per LDS stage.  Computes S^T = K * Q^T per 16x16 tile so each lane
// holds one q row with k spread over accumulator regs -> P converts to the
// WMMA A-operand layout entirely in registers (no LDS transpose).
// ---------------------------------------------------------------------------
__global__ __launch_bounds__(256)
void attn_wmma_kernel(const _Float16* __restrict__ Qh,
                      const _Float16* __restrict__ Kh,
                      const _Float16* __restrict__ VhT,   // [b][h][dv][sk]
                      const unsigned char* __restrict__ mask,
                      const float* __restrict__ aw,
                      _Float16* __restrict__ Oh)
{
#if HAVE_ASYNC_LDS
    __shared__ _Float16 Ks[2][64][72];  // ping-pong: 64 keys x 64 d (pad 8)
    __shared__ _Float16 Vs[2][64][72];  // ping-pong: 64 dv x 64 keys (pad 8)
#else
    __shared__ _Float16 Ks[1][64][72];
    __shared__ _Float16 Vs[1][64][72];
#endif

    const int blk = blockIdx.x;
    const int qc  = blk & 15;          // NQ/128 = 16 chunks
    const int h   = (blk >> 4) & 15;
    const int b   = blk >> 8;

    const int t    = threadIdx.x;
    const int w    = t >> 5;
    const int lane = t & 31;
    const int lm   = lane & 15;
    const bool hi  = lane >= 16;

    const int qbase = qc * 128 + w * 16;   // this wave's 16 q rows

    // ---- preload Q as B-operand fragments (Q^T), two K-steps of 32 ----
    const _Float16* qp = Qh + ((size_t)(b * NQ + qbase + lm)) * HD + h * 64;
    const int d0 = hi ? 8 : 0;
    AB16 qf0, qf1;
    qf0.h[0] = *(const v8h*)(qp + d0);
    qf0.h[1] = *(const v8h*)(qp + d0 + 16);
    qf1.h[0] = *(const v8h*)(qp + 32 + d0);
    qf1.h[1] = *(const v8h*)(qp + 32 + d0 + 16);

    float Mrow = -__builtin_inff();
    float Lrow = 0.0f;
    v8f O[4];
#pragma unroll
    for (int j = 0; j < 4; ++j)
#pragma unroll
        for (int i = 0; i < 8; ++i) O[j][i] = 0.0f;

    const size_t awRow = ((size_t)(b * H + h) * NQ + (qbase + lm)) * SK;

    // staging geometry: thread -> (row sr, 16-half column sc)
    const int sr = t >> 2;             // 0..63
    const int sc = (t & 3) * 16;       // 0,16,32,48
    const _Float16* kSrc = Kh  + (size_t)b * SK * HD + h * 64 +
                           (size_t)sr * HD + sc;              // + kb*HD
    const _Float16* vSrc = VhT + (size_t)((b << 4) | h) * 64 * SK +
                           (size_t)sr * SK + sc;              // + kb

#if HAVE_ASYNC_LDS
    // prologue: fill buffer 0 asynchronously
    async_cp16(kSrc,     &Ks[0][sr][sc]);
    async_cp16(kSrc + 8, &Ks[0][sr][sc + 8]);
    async_cp16(vSrc,     &Vs[0][sr][sc]);
    async_cp16(vSrc + 8, &Vs[0][sr][sc + 8]);
#else
    v8h kr0 = *(const v8h*)(kSrc);
    v8h kr1 = *(const v8h*)(kSrc + 8);
    v8h vr0 = *(const v8h*)(vSrc);
    v8h vr1 = *(const v8h*)(vSrc + 8);
#endif

    for (int kb = 0; kb < SK; kb += 64) {
#if HAVE_ASYNC_LDS
        const int buf = (kb >> 6) & 1;
        if (kb + 64 < SK) {
            // issue next tile into the other buffer, then wait for current:
            // async loads complete in order -> cnt<=4 means current buf done.
            async_cp16(kSrc + (size_t)(kb + 64) * HD,     &Ks[buf ^ 1][sr][sc]);
            async_cp16(kSrc + (size_t)(kb + 64) * HD + 8, &Ks[buf ^ 1][sr][sc + 8]);
            async_cp16(vSrc + (kb + 64),                  &Vs[buf ^ 1][sr][sc]);
            async_cp16(vSrc + (kb + 64) + 8,              &Vs[buf ^ 1][sr][sc + 8]);
            __builtin_amdgcn_s_wait_asynccnt(4);
        } else {
            __builtin_amdgcn_s_wait_asynccnt(0);
        }
        __syncthreads();
#else
        const int buf = 0;
        *(v8h*)&Ks[0][sr][sc]     = kr0;
        *(v8h*)&Ks[0][sr][sc + 8] = kr1;
        *(v8h*)&Vs[0][sr][sc]     = vr0;
        *(v8h*)&Vs[0][sr][sc + 8] = vr1;
        __syncthreads();
        if (kb + 64 < SK) {   // next tile in flight across the compute below
            kr0 = *(const v8h*)(kSrc + (size_t)(kb + 64) * HD);
            kr1 = *(const v8h*)(kSrc + (size_t)(kb + 64) * HD + 8);
            vr0 = *(const v8h*)(vSrc + (kb + 64));
            vr1 = *(const v8h*)(vSrc + (kb + 64) + 8);
        }
#endif

        // ---- S^T tiles: 4 tiles of 16 keys, K-dim = 64 (2 WMMA each) ----
        v8f S[4];
#pragma unroll
        for (int tt = 0; tt < 4; ++tt)
#pragma unroll
            for (int i = 0; i < 8; ++i) S[tt][i] = 0.0f;

        const int offA = hi ? 8 : 0;
#pragma unroll
        for (int tt = 0; tt < 4; ++tt) {
            AB16 kf;
            kf.h[0] = *(const v8h*)&Ks[buf][16 * tt + lm][offA];
            kf.h[1] = *(const v8h*)&Ks[buf][16 * tt + lm][offA + 16];
            S[tt] = WMMA16(kf.v, qf0.v, S[tt]);
            kf.h[0] = *(const v8h*)&Ks[buf][16 * tt + lm][32 + offA];
            kf.h[1] = *(const v8h*)&Ks[buf][16 * tt + lm][32 + offA + 16];
            S[tt] = WMMA16(kf.v, qf1.v, S[tt]);
        }

        // ---- scale * attn_weights (NT stream), mask, online softmax ----
        // lane holds q row (qbase+lm); tile tt reg r = key 16*tt + 8*hi + r.
        const size_t base = awRow + kb + (hi ? 8 : 0);
        if (kb + 64 < SK) {
            __builtin_prefetch(aw + base + 64, 0, 0);
            __builtin_prefetch(mask + base + 64, 0, 0);
        }

        float sv[4][8];
        float lmax = -__builtin_inff();
#pragma unroll
        for (int tt = 0; tt < 4; ++tt) {
            const v4f a0 = nt_load4(aw + base + 16 * tt);
            const v4f a1 = nt_load4(aw + base + 16 * tt + 4);
            const unsigned long long mk = nt_load_u64(mask + base + 16 * tt);
#pragma unroll
            for (int r = 0; r < 8; ++r) {
                float x = S[tt][r] * SCALE * ((r < 4) ? a0[r & 3] : a1[r & 3]);
                if ((mk >> (8 * r)) & 0xffull) x = -__builtin_inff();
                sv[tt][r] = x;
                lmax = fmaxf(lmax, x);
            }
        }
        const float rmax = fmaxf(lmax, __shfl_xor(lmax, 16, 32));
        const float Mnew = fmaxf(Mrow, rmax);
        const float fac  = __expf(Mrow - Mnew);

        float lsum = 0.0f;
        AB16 pf[2];   // P A-operand: pf[0] <- tiles 0,1 ; pf[1] <- tiles 2,3
#pragma unroll
        for (int tt = 0; tt < 4; ++tt) {
#pragma unroll
            for (int r = 0; r < 8; ++r) {
                const float p = __expf(sv[tt][r] - Mnew);
                lsum += p;
                pf[tt >> 1].v[(tt & 1) * 8 + r] = (_Float16)p;
            }
        }
        const float rsum = lsum + __shfl_xor(lsum, 16, 32);
        Lrow = Lrow * fac + rsum;
        Mrow = Mnew;

        // ---- rescale O: acc reg r belongs to q row (r + 8*hi) ----
        float fr[8];
#pragma unroll
        for (int r = 0; r < 8; ++r) fr[r] = __shfl(fac, r + (hi ? 8 : 0), 32);
#pragma unroll
        for (int j = 0; j < 4; ++j)
#pragma unroll
            for (int r = 0; r < 8; ++r) O[j][r] *= fr[r];

        // ---- O += P * V  (B-operand rows = dv, contiguous k in LDS) ----
        const int offB = hi ? 8 : 0;
#pragma unroll
        for (int j = 0; j < 4; ++j) {
            AB16 vf;
            vf.h[0] = *(const v8h*)&Vs[buf][16 * j + lm][offB];
            vf.h[1] = *(const v8h*)&Vs[buf][16 * j + lm][offB + 16];
            O[j] = WMMA16(pf[0].v, vf.v, O[j]);
            vf.h[0] = *(const v8h*)&Vs[buf][16 * j + lm][32 + offB];
            vf.h[1] = *(const v8h*)&Vs[buf][16 * j + lm][32 + offB + 16];
            O[j] = WMMA16(pf[1].v, vf.v, O[j]);
        }
        __syncthreads();
    }

    // ---- normalize and store f16 attention output (B, NQ, H*DV) ----
    const float inv = 1.0f / Lrow;
    float ir[8];
#pragma unroll
    for (int r = 0; r < 8; ++r) ir[r] = __shfl(inv, r + (hi ? 8 : 0), 32);
#pragma unroll
    for (int j = 0; j < 4; ++j) {
        const int col = h * 64 + 16 * j + lm;
#pragma unroll
        for (int r = 0; r < 8; ++r) {
            const int row = b * NQ + qbase + r + (hi ? 8 : 0);
            Oh[(size_t)row * HD + col] = (_Float16)(O[j][r] * ir[r]);
        }
    }
}

// ---------------------------------------------------------------------------
extern "C" void kernel_launch(void* const* d_in, const int* in_sizes, int n_in,
                              void* d_out, int out_size, void* d_ws, size_t ws_size,
                              hipStream_t stream)
{
    (void)in_sizes; (void)n_in; (void)out_size; (void)ws_size;

    const float*         queries  = (const float*)d_in[0];
    const float*         keys     = (const float*)d_in[1];
    const float*         values   = (const float*)d_in[2];
    const unsigned char* amask    = (const unsigned char*)d_in[3];
    const float*         aweights = (const float*)d_in[4];
    const float*         Wq       = (const float*)d_in[5];
    const float*         Wk       = (const float*)d_in[6];
    const float*         Wv       = (const float*)d_in[7];
    const float*         Wo       = (const float*)d_in[8];
    const float*         bo       = (const float*)d_in[9];
    float*               out      = (float*)d_out;

    const size_t MROWS = (size_t)BATCH * NQ;        // 4096
    _Float16* Qh  = (_Float16*)d_ws;
    _Float16* Kh  = Qh  + MROWS * HD;
    _Float16* VhT = Kh  + MROWS * HD;               // [b][h][dv][sk]
    _Float16* Oh  = VhT + MROWS * HD;

    const dim3 gGemm(DMODEL / 128, (unsigned)(MROWS / 128));   // (8, 32)

    // Q/K projections: f32 in, f16 row-major out
    gemm_wmma_kernel<0><<<gGemm, 256, 0, stream>>>(
        queries, Wq, nullptr, Qh, (int)MROWS, DMODEL, DMODEL);
    gemm_wmma_kernel<0><<<gGemm, 256, 0, stream>>>(
        keys, Wk, nullptr, Kh, (int)MROWS, DMODEL, DMODEL);
    // V projection: f16 out, transposed [b][h][dv][sk]
    gemm_wmma_kernel<1><<<gGemm, 256, 0, stream>>>(
        values, Wv, nullptr, VhT, (int)MROWS, DMODEL, DMODEL);

    // Flash attention: 512 blocks of 256 threads
    attn_wmma_kernel<<<BATCH * H * (NQ / 128), 256, 0, stream>>>(
        Qh, Kh, VhT, amask, aweights, Oh);

    // Output projection: f16 in, f32 out + bias (non-temporal)
    gemm_wmma_kernel<2><<<gGemm, 256, 0, stream>>>(
        Oh, Wo, bo, out, (int)MROWS, DMODEL, DMODEL);
}
